// BertAttention_78752520339946
// MI455X (gfx1250) — matrix operand order repaired and checked by reading, not compiled, not run on hardware
//
#include <hip/hip_runtime.h>

// ---------------------------------------------------------------------------
// BERT attention w/ dual KV caches on MI455X (gfx1250, wave32, WMMA bf16)
//   q,k,v = hidden@W+b ; scores = [q·K0(b%8)ᵀ | q·K1ᵀ | q·Kselfᵀ]/8 + mask
//   probs = softmax(scores) ; ctx = probs @ [V0|V1|Vself]
// Strategy: bf16 operands, f32 WMMA accumulation (v_wmma_f32_16x16x32_bf16),
// double-buffered GLOBAL_LOAD_ASYNC_TO_LDS_B128 staging (ASYNCcnt-tracked).
// ---------------------------------------------------------------------------

typedef __bf16 bf16_t;
typedef __attribute__((ext_vector_type(16))) __bf16 v16bf;
typedef __attribute__((ext_vector_type(8)))  __bf16 v8bf;
typedef __attribute__((ext_vector_type(8)))  float  v8f;

#define NUM_HEADS 12
#define HEAD_DIM  64
#define HIDDEN    768
#define BSZ       240          // FULL
#define SEQ       32           // SLOT_LEN
#define LEN0      512
#define LEN1      128
#define KTOT      (LEN0 + LEN1 + SEQ)   // 672
#define SLD       680          // padded LDS leading dim for scores/probs

// ---------------- CDNA5 async global->LDS copy (ASYNCcnt) ------------------

// 16B DMA: memory -> LDS, tracked by ASYNCcnt (CDNA5 ISA).
// Low 32 bits of a generic LDS pointer are the wave-relative LDS offset.
__device__ __forceinline__ void async_cp16(const bf16_t* g, bf16_t* l) {
    unsigned lds = (unsigned)(size_t)l;
    unsigned long long ga = (unsigned long long)(size_t)g;
    asm volatile("global_load_async_to_lds_b128 %0, %1, off"
                 :: "v"(lds), "v"(ga) : "memory");
}

#if __has_builtin(__builtin_amdgcn_s_wait_asynccnt)
#define WAIT_ASYNC(n) __builtin_amdgcn_s_wait_asynccnt(n)
#else
#define WAIT_ASYNC(n) asm volatile("s_wait_asynccnt %0" :: "i"(n) : "memory")
#endif

// ---- CDNA5 WMMA fragment helpers (wave32 layouts, CDNA5 ISA) ----

// A (16x32 bf16, MxK): lane m=lane&15; lanes>=16 take K-offset +8.
// VGPR0-3 <- K = kb..kb+7 ; VGPR4-7 <- K = 16+kb..16+kb+7 (contiguous runs).
__device__ __forceinline__ v16bf frag_a(const bf16_t* p, int ld, int lane) {
    int m  = lane & 15;
    int kb = (lane >> 4) << 3;              // 0 or 8
    const bf16_t* q = p + m * ld + kb;
    v8bf lo = *(const v8bf*)(q);            // K = kb .. kb+7
    v8bf hi = *(const v8bf*)(q + 16);       // K = 16+kb .. 16+kb+7
    return __builtin_shufflevector(lo, hi, 0,1,2,3,4,5,6,7,8,9,10,11,12,13,14,15);
}

// B (32x16 bf16, KxN) given B^T row-major (BT[n][k]): lane n=lane&15;
// lanes>=16 take K-offset +8; VGPR0-3 <- K = kb..kb+7, VGPR4-7 <- K = 16+kb..16+kb+7.
__device__ __forceinline__ v16bf frag_b(const bf16_t* bt, int ld, int lane) {
    int n  = lane & 15;
    int kb = (lane >> 4) << 3;              // 0 or 8
    const bf16_t* q = bt + n * ld + kb;
    v8bf lo = *(const v8bf*)(q);
    v8bf hi = *(const v8bf*)(q + 16);
    return __builtin_shufflevector(lo, hi, 0,1,2,3,4,5,6,7,8,9,10,11,12,13,14,15);
}

__device__ __forceinline__ v8f wmma_bf16(v16bf a, v16bf b, v8f c) {
    // (neg_a, A, neg_b, B, c_mod, C, reuse_a, reuse_b)
    return __builtin_amdgcn_wmma_f32_16x16x32_bf16(false, a, false, b, (short)0, c, false, false);
}

__device__ __forceinline__ v8f zero8() {
    v8f z = {0.f,0.f,0.f,0.f,0.f,0.f,0.f,0.f};
    return z;
}

// ---------------------------- convert kernels ------------------------------

__global__ __launch_bounds__(256) void cvt_bf16(const float* __restrict__ s,
                                                bf16_t* __restrict__ d, int n) {
    int i = blockIdx.x * 256 + threadIdx.x;
    if (i < n) d[i] = (bf16_t)s[i];
}

// dst[m][c][r] = src[m][r][c]  (per-matrix transpose + f32->bf16)
__global__ __launch_bounds__(256) void cvt_transpose_bf16(const float* __restrict__ s,
                                                          bf16_t* __restrict__ d,
                                                          int RC, int R, int C, int n) {
    int i = blockIdx.x * 256 + threadIdx.x;
    if (i < n) {
        int m = i / RC;
        int rem = i - m * RC;
        int c = rem / R;
        int r = rem - c * R;
        d[i] = (bf16_t)s[(size_t)m * RC + (size_t)r * C + c];
    }
}

// ------------------------------ QKV GEMM -----------------------------------
// C(7680x768) = X(7680x768) @ W ; W supplied transposed (WT[n][k]).
// Block tile 64x64, 4 waves of 32x32, double-buffered async LDS staging.
// z selects {Q,K,V}. Q,K stored split-head (B,H,L,64) bf16; V stored
// transposed (B,H,64,L) bf16 for the P·V GEMM.
__global__ __launch_bounds__(128) void qkv_gemm(
        const bf16_t* __restrict__ X,
        const bf16_t* __restrict__ WT0, const bf16_t* __restrict__ WT1,
        const bf16_t* __restrict__ WT2,
        const float* __restrict__ bias0, const float* __restrict__ bias1,
        const float* __restrict__ bias2,
        bf16_t* __restrict__ Qo, bf16_t* __restrict__ Ko, bf16_t* __restrict__ Vo) {
    __shared__ bf16_t As[2][64 * 40];   // 64 rows x 32 k, padded ld=40
    __shared__ bf16_t Bs[2][64 * 40];   // 64 cols x 32 k (WT rows), padded ld=40

    const int z = blockIdx.z;
    const bf16_t* WT  = (z == 0) ? WT0 : ((z == 1) ? WT1 : WT2);
    const float* bias = (z == 0) ? bias0 : ((z == 1) ? bias1 : bias2);
    bf16_t* dst       = (z == 0) ? Qo : ((z == 1) ? Ko : Vo);

    const int row0 = blockIdx.y * 64;
    const int col0 = blockIdx.x * 64;
    const int tid  = threadIdx.x;
    const int lane = tid & 31;
    const int wv   = tid >> 5;
    const int wm   = wv >> 1, wn = wv & 1;

    // per-thread staging slots: chunks tid and tid+128 of 256 (16B each)
    const int r0 = tid >> 2,         o0 = (tid & 3) * 8;
    const int r1 = (tid + 128) >> 2, o1 = ((tid + 128) & 3) * 8;
    const bf16_t* XA0 = X  + (size_t)(row0 + r0) * HIDDEN + o0;
    const bf16_t* XA1 = X  + (size_t)(row0 + r1) * HIDDEN + o1;
    const bf16_t* WB0 = WT + (size_t)(col0 + r0) * HIDDEN + o0;
    const bf16_t* WB1 = WT + (size_t)(col0 + r1) * HIDDEN + o1;

    auto issue = [&](int step, int buf) {
        int k0 = step * 32;
        async_cp16(XA0 + k0, &As[buf][r0 * 40 + o0]);
        async_cp16(XA1 + k0, &As[buf][r1 * 40 + o1]);
        async_cp16(WB0 + k0, &Bs[buf][r0 * 40 + o0]);
        async_cp16(WB1 + k0, &Bs[buf][r1 * 40 + o1]);
    };

    v8f acc[4] = {zero8(), zero8(), zero8(), zero8()};

    issue(0, 0);
    for (int i = 0; i < HIDDEN / 32; ++i) {
        int buf = i & 1;
        if (i + 1 < HIDDEN / 32) {       // prefetch next stage into other buffer
            issue(i + 1, buf ^ 1);
            WAIT_ASYNC(4);               // step-i loads (oldest 4) complete
        } else {
            WAIT_ASYNC(0);
        }
        __syncthreads();                 // publish buf to all waves
        v16bf a0 = frag_a(&As[buf][(wm * 32     ) * 40], 40, lane);
        v16bf a1 = frag_a(&As[buf][(wm * 32 + 16) * 40], 40, lane);
        v16bf b0 = frag_b(&Bs[buf][(wn * 32     ) * 40], 40, lane);
        v16bf b1 = frag_b(&Bs[buf][(wn * 32 + 16) * 40], 40, lane);
        acc[0] = wmma_bf16(a0, b0, acc[0]);
        acc[1] = wmma_bf16(a0, b1, acc[1]);
        acc[2] = wmma_bf16(a1, b0, acc[2]);
        acc[3] = wmma_bf16(a1, b1, acc[3]);
        __syncthreads();                 // all reads done before buf overwrite
    }

    const int n_ = lane & 15;
    const int mb = (lane >> 4) << 3;
    #pragma unroll
    for (int t = 0; t < 4; ++t) {
        int mt = t >> 1, nt = t & 1;
        int gcol = col0 + wn * 32 + nt * 16 + n_;
        float bv = bias[gcol];
        int hh = gcol >> 6, d = gcol & 63;
        #pragma unroll
        for (int r = 0; r < 8; ++r) {
            int grow = row0 + wm * 32 + mt * 16 + mb + r;
            int bb = grow >> 5, l = grow & 31;
            float val = acc[t][r] + bv;
            if (z < 2)
                dst[(((size_t)bb * NUM_HEADS + hh) * SEQ + l) * HEAD_DIM + d] = (bf16_t)val;
            else
                dst[(((size_t)bb * NUM_HEADS + hh) * HEAD_DIM + d) * SEQ + l] = (bf16_t)val;
        }
    }
}

// ------------------------------ Attention ----------------------------------
// One block per (b, head). 128 threads = 4 waves.
__global__ __launch_bounds__(128) void attention(
        const bf16_t* __restrict__ Qbf,  const bf16_t* __restrict__ Kbf,
        const bf16_t* __restrict__ VTbf,
        const bf16_t* __restrict__ c0k,  const bf16_t* __restrict__ c0vT,
        const bf16_t* __restrict__ c1k,  const bf16_t* __restrict__ c1vT,
        const float*  __restrict__ mask, float* __restrict__ out) {
    __shared__ bf16_t Qs[SEQ * 72];       // 32x64 Q tile, padded ld=72
    __shared__ float  Ss[SEQ * SLD];      // f32 scores 32x672 (padded)
    __shared__ bf16_t Ps[SEQ * SLD];      // bf16 probs  32x672 (padded)
    __shared__ float  red[SEQ][4];

    const int bh = blockIdx.x;            // 0..2879
    const int b  = bh / NUM_HEADS;
    const int h  = bh - b * NUM_HEADS;
    const int tid  = threadIdx.x;
    const int lane = tid & 31;
    const int wv   = tid >> 5;

    // stage Q tile via async DMA (2 x 16B chunks per thread)
    {
        const bf16_t* Qg = Qbf + (size_t)bh * (SEQ * HEAD_DIM);
        int ra = tid >> 3,         oa = (tid & 7) * 8;
        int rb = (tid + 128) >> 3, ob = ((tid + 128) & 7) * 8;
        async_cp16(Qg + ra * HEAD_DIM + oa, &Qs[ra * 72 + oa]);
        async_cp16(Qg + rb * HEAD_DIM + ob, &Qs[rb * 72 + ob]);
        WAIT_ASYNC(0);
    }
    __syncthreads();

    // ---- scores = Q @ K^T for 42 column tiles of 16 keys -------------------
    const bf16_t* Ks0 = c0k + (size_t)((b & 7) * NUM_HEADS + h) * (LEN0 * HEAD_DIM);
    const bf16_t* Ks1 = c1k + (size_t)bh * (LEN1 * HEAD_DIM);
    const bf16_t* Kss = Kbf + (size_t)bh * (SEQ  * HEAD_DIM);

    for (int nt = wv; nt < 42; nt += 4) {
        const bf16_t* kb_;
        int rb;
        if (nt < 32)      { kb_ = Ks0; rb = nt * 16; }
        else if (nt < 40) { kb_ = Ks1; rb = (nt - 32) * 16; }
        else              { kb_ = Kss; rb = (nt - 40) * 16; }
        const bf16_t* bt = kb_ + (size_t)rb * HEAD_DIM;

        v8f acc0 = zero8(), acc1 = zero8();
        #pragma unroll
        for (int k0 = 0; k0 < HEAD_DIM; k0 += 32) {
            v16bf fb = frag_b(bt + k0, HEAD_DIM, lane);       // keys direct from global
            v16bf a0 = frag_a(&Qs[0 * 72 + k0], 72, lane);
            v16bf a1 = frag_a(&Qs[16 * 72 + k0], 72, lane);
            acc0 = wmma_bf16(a0, fb, acc0);
            acc1 = wmma_bf16(a1, fb, acc1);
        }
        int n = lane & 15, mb = (lane >> 4) << 3;
        #pragma unroll
        for (int r = 0; r < 8; ++r) {
            Ss[(mb + r)      * SLD + nt * 16 + n] = acc0[r];
            Ss[(16 + mb + r) * SLD + nt * 16 + n] = acc1[r];
        }
    }
    __syncthreads();

    // ---- softmax over 672 (scale 1/8 + mask) ------------------------------
    const int row  = tid >> 2;
    const int part = tid & 3;
    const int cc0  = part * (KTOT / 4);    // 168 columns per thread
    const float* mrow = mask + (size_t)b * KTOT;

    float mx = -3.4e38f;
    for (int c = 0; c < KTOT / 4; ++c) {
        float s = Ss[row * SLD + cc0 + c] * 0.125f + mrow[cc0 + c];
        Ss[row * SLD + cc0 + c] = s;
        mx = fmaxf(mx, s);
    }
    red[row][part] = mx;
    __syncthreads();
    mx = fmaxf(fmaxf(red[row][0], red[row][1]), fmaxf(red[row][2], red[row][3]));

    float sm = 0.f;
    for (int c = 0; c < KTOT / 4; ++c) {
        float e = __expf(Ss[row * SLD + cc0 + c] - mx);
        Ss[row * SLD + cc0 + c] = e;
        sm += e;
    }
    red[row][part] = sm;
    __syncthreads();
    float inv = 1.f / (red[row][0] + red[row][1] + red[row][2] + red[row][3]);
    for (int c = 0; c < KTOT / 4; ++c)
        Ps[row * SLD + cc0 + c] = (bf16_t)(Ss[row * SLD + cc0 + c] * inv);
    __syncthreads();

    // ---- ctx = P @ V  (V supplied transposed: VT[d][kpos]) -----------------
    const bf16_t* V0 = c0vT + (size_t)((b & 7) * NUM_HEADS + h) * (HEAD_DIM * LEN0);
    const bf16_t* V1 = c1vT + (size_t)bh * (HEAD_DIM * LEN1);
    const bf16_t* Vs = VTbf + (size_t)bh * (HEAD_DIM * SEQ);

    for (int t = wv; t < 8; t += 4) {       // 2 m-tiles x 4 d-tiles
        int mt = t >> 2, ntt = t & 3;
        v8f acc = zero8();
        for (int ks = 0; ks < KTOT / 32; ++ks) {
            int kp = ks * 32;
            const bf16_t* vt; int ldv, loc;
            if (kp < LEN0)             { vt = V0; ldv = LEN0; loc = kp; }
            else if (kp < LEN0 + LEN1) { vt = V1; ldv = LEN1; loc = kp - LEN0; }
            else                       { vt = Vs; ldv = SEQ;  loc = kp - LEN0 - LEN1; }
            v16bf fb = frag_b(vt + (size_t)(ntt * 16) * ldv + loc, ldv, lane);
            v16bf fa = frag_a(&Ps[(mt * 16) * SLD + kp], SLD, lane);
            acc = wmma_bf16(fa, fb, acc);
        }
        int n = lane & 15, mb = (lane >> 4) << 3;
        int d = h * HEAD_DIM + ntt * 16 + n;
        #pragma unroll
        for (int r = 0; r < 8; ++r) {
            int l = mt * 16 + mb + r;
            out[((size_t)(b * SEQ + l)) * HIDDEN + d] = acc[r];
        }
    }
}

// ------------------------------- launcher ----------------------------------

extern "C" void kernel_launch(void* const* d_in, const int* in_sizes, int n_in,
                              void* d_out, int out_size, void* d_ws, size_t ws_size,
                              hipStream_t stream) {
    const float* hidden = (const float*)d_in[0];
    const float* mask   = (const float*)d_in[1];
    const float* c0k    = (const float*)d_in[2];
    const float* c0v    = (const float*)d_in[3];
    const float* c1k    = (const float*)d_in[4];
    const float* c1v    = (const float*)d_in[5];
    const float* Wq     = (const float*)d_in[6];
    const float* bq     = (const float*)d_in[7];
    const float* Wk     = (const float*)d_in[8];
    const float* bk     = (const float*)d_in[9];
    const float* Wv     = (const float*)d_in[10];
    const float* bv     = (const float*)d_in[11];
    float* out = (float*)d_out;

    char* ws = (char*)d_ws;
    size_t off = 0;
    auto carve = [&](size_t elems) -> bf16_t* {
        bf16_t* p = (bf16_t*)(ws + off);
        off += ((elems * sizeof(bf16_t) + 255) / 256) * 256;
        return p;
    };

    const int nHid  = BSZ * SEQ * HIDDEN;                 // 5,898,240
    const int nW    = HIDDEN * HIDDEN;                    //   589,824
    const int nQKV  = BSZ * NUM_HEADS * SEQ * HEAD_DIM;   // 5,898,240
    const int nC0   = 8 * NUM_HEADS * LEN0 * HEAD_DIM;    // 3,145,728
    const int nC1   = BSZ * NUM_HEADS * LEN1 * HEAD_DIM;  // 23,592,960

    bf16_t* hid_bf = carve(nHid);
    bf16_t* WTq    = carve(nW);
    bf16_t* WTk    = carve(nW);
    bf16_t* WTv    = carve(nW);
    bf16_t* Qbf    = carve(nQKV);
    bf16_t* Kbf    = carve(nQKV);
    bf16_t* VTbf   = carve(nQKV);
    bf16_t* c0k_bf = carve(nC0);
    bf16_t* c0vT   = carve(nC0);
    bf16_t* c1k_bf = carve(nC1);
    bf16_t* c1vT   = carve(nC1);
    (void)ws_size; (void)in_sizes; (void)n_in; (void)out_size;

    auto grid1 = [](int n) { return dim3((unsigned)((n + 255) / 256)); };

    cvt_bf16<<<grid1(nHid), 256, 0, stream>>>(hidden, hid_bf, nHid);
    cvt_bf16<<<grid1(nC0),  256, 0, stream>>>(c0k, c0k_bf, nC0);
    cvt_bf16<<<grid1(nC1),  256, 0, stream>>>(c1k, c1k_bf, nC1);
    cvt_transpose_bf16<<<grid1(nW), 256, 0, stream>>>(Wq, WTq, nW, HIDDEN, HIDDEN, nW);
    cvt_transpose_bf16<<<grid1(nW), 256, 0, stream>>>(Wk, WTk, nW, HIDDEN, HIDDEN, nW);
    cvt_transpose_bf16<<<grid1(nW), 256, 0, stream>>>(Wv, WTv, nW, HIDDEN, HIDDEN, nW);
    cvt_transpose_bf16<<<grid1(nC0), 256, 0, stream>>>(c0v, c0vT, LEN0 * HEAD_DIM, LEN0, HEAD_DIM, nC0);
    cvt_transpose_bf16<<<grid1(nC1), 256, 0, stream>>>(c1v, c1vT, LEN1 * HEAD_DIM, LEN1, HEAD_DIM, nC1);

    dim3 ggrid(HIDDEN / 64, (BSZ * SEQ) / 64, 3);  // 12 x 120 x 3
    qkv_gemm<<<ggrid, 128, 0, stream>>>(hid_bf, WTq, WTk, WTv, bq, bk, bv, Qbf, Kbf, VTbf);

    attention<<<dim3(BSZ * NUM_HEADS), 128, 0, stream>>>(
        Qbf, Kbf, VTbf, c0k_bf, c0vT, c1k_bf, c1vT, mask, out);
}
